// Llama4TextAttention_37589553775177
// MI455X (gfx1250) — matrix operand order repaired and checked by reading, not compiled, not run on hardware
//
#include <hip/hip_runtime.h>
#include <hip/hip_bf16.h>

// ---------------------------------------------------------------------------
// Llama4TextAttention for MI455X (gfx1250), bf16 WMMA pipeline.
//   B=1, S=2048, D=4096, H=32, KVH=8, HD=128, N_REP=4
// Pipeline:
//   1. f32 -> bf16 conversion of X and all weights (one pass, L2-resident after)
//   2. QKV projections: bf16 WMMA GEMM with double-buffered
//      GLOBAL_LOAD_ASYNC_TO_LDS_B128 tile staging (ASYNCcnt pipeline)
//   3. RoPE + L2-norm fused elementwise -> bf16, head-major layout
//   4. V transposed to [kh][d][s] bf16 for contiguous P@V B-fragments
//   5. Flash attention: one wave per (16-row q block, head), online softmax,
//      v_wmma_f32_16x16x32_bf16 for both Q@K^T and P@V. Causal mask computed
//      analytically (the 16MB mask input is never read).
//   6. Output projection: same WMMA GEMM -> f32 d_out
// ---------------------------------------------------------------------------

#define S_LEN    2048
#define D_MODEL  4096
#define NH_Q     32
#define NH_KV    8
#define HEAD_DIM 128
#define NREP     4
#define NEG_BIG  (-1.0e9f)
#define ATT_SCALE 0.08838834764831845f   // 128^-0.5

typedef __bf16 bf16_t;
typedef __attribute__((ext_vector_type(16))) __bf16 v16bf;
typedef __attribute__((ext_vector_type(8)))  __bf16 v8bf;
typedef __attribute__((ext_vector_type(8)))  float  v8f;

// --------------------------- WMMA helpers ----------------------------------

__device__ __forceinline__ v8f wmma_bf16(v16bf a, v16bf b, v8f c) {
  // D = A(16x32) * B(32x16) + C, f32 accumulate
  return __builtin_amdgcn_wmma_f32_16x16x32_bf16(
      /*neg_a=*/false, a, /*neg_b=*/false, b,
      /*c_mod=*/(short)0, c, /*reuse_a=*/false, /*reuse_b=*/false);
}

// A-matrix fragment (16x32, 16-bit): per ISA 7.12.2, lane L holds row L%16;
// vector elems 0..7 = K (half? 8..15 : 0..7), elems 8..15 = K (half? 24..31 : 16..23).
// Source matrix stored row-major (rows, K) with leading dim ld.
__device__ __forceinline__ v16bf load_frag_a(const bf16_t* __restrict__ p,
                                             int row0, int k0, int ld) {
  int lane = threadIdx.x & 31;
  int r    = lane & 15;
  int koff = (lane & 16) ? 8 : 0;
  const bf16_t* rp = p + (size_t)(row0 + r) * ld + k0 + koff;
  union { v16bf v; v8bf h[2]; } u;
  u.h[0] = *(const v8bf*)(rp);        // K = k0+koff .. +7
  u.h[1] = *(const v8bf*)(rp + 16);   // K = k0+koff+16 .. +23
  return u.v;
}

// B-matrix fragment (32x16, 16-bit): lane L holds column L%16;
// elems 0..7 = K (half? 8..15 : 0..7), elems 8..15 = K (half? 24..31 : 16..23).
// Source stored row-major (cols, K): B[k][n] = src[n][k]  (i.e. X @ W^T form).
__device__ __forceinline__ v16bf load_frag_b(const bf16_t* __restrict__ p,
                                             int col0, int k0, int ld) {
  int lane = threadIdx.x & 31;
  int n    = lane & 15;
  int koff = (lane & 16) ? 8 : 0;
  const bf16_t* rp = p + (size_t)(col0 + n) * ld + k0 + koff;
  union { v16bf v; v8bf h[2]; } u;
  u.h[0] = *(const v8bf*)(rp);        // K = k0+koff    .. +7
  u.h[1] = *(const v8bf*)(rp + 16);   // K = k0+koff+16 .. +23
  return u.v;
}

// Async global->LDS copy of one 16-byte chunk per lane (CDNA5 TENSOR/ASYNC
// data path, tracked by ASYNCcnt). LDS address = low 32 bits of the generic
// pointer to the __shared__ destination (LDS aperture truncation, ISA §10.2).
__device__ __forceinline__ void async_copy_b128(bf16_t* lds_dst,
                                                const bf16_t* gsrc) {
  unsigned           loff = (unsigned)(size_t)lds_dst;
  unsigned long long gadr = (unsigned long long)(size_t)gsrc;
  asm volatile("global_load_async_to_lds_b128 %0, %1, off"
               :: "v"(loff), "v"(gadr) : "memory");
}

__device__ __forceinline__ void wait_async_le3() {
  asm volatile("s_wait_asynccnt 0x3" ::: "memory");
}
__device__ __forceinline__ void wait_async_0() {
  asm volatile("s_wait_asynccnt 0x0" ::: "memory");
}

// --------------------------- elementwise kernels ---------------------------

__global__ void f32_to_bf16_kernel(const float* __restrict__ src,
                                   bf16_t* __restrict__ dst, size_t n) {
  size_t i = (size_t)blockIdx.x * blockDim.x + threadIdx.x;
  if (i < n) dst[i] = (bf16_t)src[i];
}

// RoPE + L2-norm, fused. src: f32 [S, NH*HD] (projection output layout).
// dst: bf16 [NH][S][HD] (head-major for attention).
__global__ void rope_l2norm_kernel(const float* __restrict__ src,
                                   const float* __restrict__ cosp,
                                   const float* __restrict__ sinp,
                                   bf16_t* __restrict__ dst, int NH) {
  int s = blockIdx.x;      // sequence position
  int h = blockIdx.y;      // head
  int t = threadIdx.x;     // pair index 0..63
  const float* row = src + (size_t)s * NH * HEAD_DIM + (size_t)h * HEAD_DIM;
  float x0 = row[2 * t];
  float x1 = row[2 * t + 1];
  float c  = cosp[s * (HEAD_DIM / 2) + t];
  float sn = sinp[s * (HEAD_DIM / 2) + t];
  float re = x0 * c - x1 * sn;
  float im = x0 * sn + x1 * c;

  __shared__ float red[64];
  red[t] = re * re + im * im;
  __syncthreads();
  for (int off = 32; off > 0; off >>= 1) {
    if (t < off) red[t] += red[t + off];
    __syncthreads();
  }
  float scale = rsqrtf(red[0] * (1.0f / HEAD_DIM) + 1e-5f);

  bf16_t* drow = dst + ((size_t)h * S_LEN + s) * HEAD_DIM;
  drow[2 * t]     = (bf16_t)(re * scale);
  drow[2 * t + 1] = (bf16_t)(im * scale);
}

// V: f32 [S, KVH*HD] -> bf16 [KVH][HD][S] (transposed, d-major rows).
__global__ void v_transpose_kernel(const float* __restrict__ src,
                                   bf16_t* __restrict__ dst) {
  size_t idx = (size_t)blockIdx.x * blockDim.x + threadIdx.x;
  // idx -> (s, kh, d)
  int d  = (int)(idx % HEAD_DIM);
  int kh = (int)((idx / HEAD_DIM) % NH_KV);
  int s  = (int)(idx / (HEAD_DIM * NH_KV));
  if (s >= S_LEN) return;
  float v = src[idx];
  dst[((size_t)kh * HEAD_DIM + d) * S_LEN + s] = (bf16_t)v;
}

// --------------------------- GEMM: C = A @ B^T -----------------------------
// A: bf16 (M,K) row-major. B: bf16 (N,K) row-major. C: f32 (M,N) row-major.
// Block = 256 threads = 8 waves (2 M x 4 N), block tile 64(M) x 128(N),
// wave tile 32x32 (2x2 WMMA). Per K-step of 32, the 64x32 A tile (4KB) and
// 128x32 B tile (8KB) are staged into LDS with GLOBAL_LOAD_ASYNC_TO_LDS_B128,
// double-buffered: issue batch k+1 (3 async ops/wave), s_wait_asynccnt<=3
// guarantees batch k landed (async loads complete in order), barrier, compute.
__global__ __launch_bounds__(256)
void gemm_bf16_kernel(const bf16_t* __restrict__ A, const bf16_t* __restrict__ B,
                      float* __restrict__ C, int M, int N, int K) {
  __shared__ __attribute__((aligned(16))) bf16_t lA[2][64][32];
  __shared__ __attribute__((aligned(16))) bf16_t lB[2][128][32];

  int tid  = threadIdx.x;
  int lane = tid & 31;
  int wid  = tid >> 5;
  int wm   = wid & 1;
  int wn   = wid >> 1;
  int blockRow = blockIdx.y * 64;
  int blockCol = blockIdx.x * 128;

  // async-copy assignments: A = 64 rows x 4 chunks (1/thread),
  // B = 128 cols x 4 chunks (2/thread); chunk = 8 bf16 = 16 bytes.
  int aRow = tid >> 2, aChk = tid & 3;
  int bI0  = tid * 2,  bI1  = tid * 2 + 1;
  int bCol0 = bI0 >> 2, bChk0 = bI0 & 3;
  int bCol1 = bI1 >> 2, bChk1 = bI1 & 3;
  const bf16_t* gA  = A + (size_t)(blockRow + aRow)  * K + aChk  * 8;
  const bf16_t* gB0 = B + (size_t)(blockCol + bCol0) * K + bChk0 * 8;
  const bf16_t* gB1 = B + (size_t)(blockCol + bCol1) * K + bChk1 * 8;

  v8f acc00 = {}, acc01 = {}, acc10 = {}, acc11 = {};

  int nsteps = K >> 5;

  // prologue: batch 0 into buffer 0
  async_copy_b128(&lA[0][aRow][aChk * 8],   gA);
  async_copy_b128(&lB[0][bCol0][bChk0 * 8], gB0);
  async_copy_b128(&lB[0][bCol1][bChk1 * 8], gB1);

  for (int s = 0; s < nsteps; ++s) {
    int buf = s & 1;
    if (s + 1 < nsteps) {
      int k1 = (s + 1) << 5;
      async_copy_b128(&lA[buf ^ 1][aRow][aChk * 8],   gA  + k1);
      async_copy_b128(&lB[buf ^ 1][bCol0][bChk0 * 8], gB0 + k1);
      async_copy_b128(&lB[buf ^ 1][bCol1][bChk1 * 8], gB1 + k1);
      wait_async_le3();            // batch s complete (in-order completion)
    } else {
      wait_async_0();
    }
    __syncthreads();               // all waves' async writes landed

    const bf16_t* pA = &lA[buf][0][0];
    const bf16_t* pB = &lB[buf][0][0];
    v16bf a0 = load_frag_a(pA, wm * 32,      0, 32);
    v16bf a1 = load_frag_a(pA, wm * 32 + 16, 0, 32);
    v16bf b0 = load_frag_b(pB, wn * 32,      0, 32);
    v16bf b1 = load_frag_b(pB, wn * 32 + 16, 0, 32);
    acc00 = wmma_bf16(a0, b0, acc00);
    acc01 = wmma_bf16(a0, b1, acc01);
    acc10 = wmma_bf16(a1, b0, acc10);
    acc11 = wmma_bf16(a1, b1, acc11);

    __syncthreads();               // protect buf reuse (WAR vs next issue)
  }

  int hgrp = lane >> 4;   // lane-half: rows r / r+8 (C/D layout)
  int l16  = lane & 15;   // column within 16-wide tile
  int rowBase = blockRow + wm * 32;
  int colBase = blockCol + wn * 32;
#pragma unroll
  for (int r = 0; r < 8; ++r) {
    int row0 = rowBase + r + hgrp * 8;
    int row1 = row0 + 16;
    C[(size_t)row0 * N + colBase + l16]      = acc00[r];
    C[(size_t)row0 * N + colBase + 16 + l16] = acc01[r];
    C[(size_t)row1 * N + colBase + l16]      = acc10[r];
    C[(size_t)row1 * N + colBase + 16 + l16] = acc11[r];
  }
}

// --------------------------- Flash attention -------------------------------
// One wave per (16-query block, head). Online softmax, causal.
// Qb: bf16 [H][S][HD], Kb: bf16 [KVH][S][HD], Vt: bf16 [KVH][HD][S].
// Ctx: bf16 [S][H*HD] (s-major for the output projection GEMM).
__global__ __launch_bounds__(32)
void attn_kernel(const bf16_t* __restrict__ Qb, const bf16_t* __restrict__ Kb,
                 const bf16_t* __restrict__ Vt, bf16_t* __restrict__ Ctx) {
  int qblk = blockIdx.x;         // 16-row query block
  int h    = blockIdx.y;         // query head
  int kh   = h / NREP;           // kv head
  int lane = threadIdx.x & 31;
  int hgrp = lane >> 4;          // 0: rows 0..7, 1: rows 8..15 (C/D layout)
  int l16  = lane & 15;

  const bf16_t* Qh = Qb + (size_t)h  * S_LEN * HEAD_DIM;
  const bf16_t* Kh = Kb + (size_t)kh * S_LEN * HEAD_DIM;
  const bf16_t* Vh = Vt + (size_t)kh * HEAD_DIM * S_LEN;
  int q0 = qblk * 16;

  // Q fragments over the 128-wide head dim (4 K-steps of 32)
  v16bf qf[4];
#pragma unroll
  for (int d = 0; d < 4; ++d) qf[d] = load_frag_a(Qh, q0, d * 32, HEAD_DIM);

  v8f acc[8];
#pragma unroll
  for (int g = 0; g < 8; ++g) acc[g] = (v8f){};
  float mrow[8], lrow[8];
#pragma unroll
  for (int r = 0; r < 8; ++r) { mrow[r] = -3.0e38f; lrow[r] = 0.0f; }

  __shared__ __attribute__((aligned(16))) bf16_t plds[16 * 32]; // P tile

  int nkb = (q0 + 16 + 31) >> 5;     // causal: key blocks of 32 covering q0+15
  for (int kb = 0; kb < nkb; ++kb) {
    int key0 = kb * 32;

    // scores: S(16x32) = Q(16x128) @ K^T, two 16x16 WMMA tiles
    v8f s0 = {}, s1 = {};
#pragma unroll
    for (int d = 0; d < 4; ++d) {
      v16bf b0 = load_frag_b(Kh, key0,      d * 32, HEAD_DIM);
      v16bf b1 = load_frag_b(Kh, key0 + 16, d * 32, HEAD_DIM);
      s0 = wmma_bf16(qf[d], b0, s0);
      s1 = wmma_bf16(qf[d], b1, s1);
    }

    int kn0 = key0 + l16;        // key index for tile 0 column
    int kn1 = key0 + 16 + l16;   // key index for tile 1 column
#pragma unroll
    for (int r = 0; r < 8; ++r) {
      int m    = r + hgrp * 8;
      int qidx = q0 + m;
      float v0 = s0[r] * ATT_SCALE + ((kn0 <= qidx) ? 0.0f : NEG_BIG);
      float v1 = s1[r] * ATT_SCALE + ((kn1 <= qidx) ? 0.0f : NEG_BIG);

      // row max across the 16 lanes of this half (xor 1/2/4/8 stays in-half)
      float bm = fmaxf(v0, v1);
#pragma unroll
      for (int off = 1; off < 16; off <<= 1)
        bm = fmaxf(bm, __shfl_xor(bm, off, 32));

      float mnew = fmaxf(mrow[r], bm);
      float corr = __expf(mrow[r] - mnew);
      float p0   = __expf(v0 - mnew);
      float p1   = __expf(v1 - mnew);
      float rs   = p0 + p1;
#pragma unroll
      for (int off = 1; off < 16; off <<= 1)
        rs += __shfl_xor(rs, off, 32);

      lrow[r] = lrow[r] * corr + rs;
      mrow[r] = mnew;
#pragma unroll
      for (int g = 0; g < 8; ++g) acc[g][r] *= corr;

      // stash P (C/D layout -> LDS row-major) for A-fragment re-read
      plds[m * 32 + l16]      = (bf16_t)p0;
      plds[m * 32 + 16 + l16] = (bf16_t)p1;
    }
    __syncthreads();

    // re-read P in A-fragment layout (contiguous 8-elem chunks)
    int koff = hgrp ? 8 : 0;
    union { v16bf v; v8bf hv[2]; } pu;
    pu.hv[0] = *(const v8bf*)&plds[l16 * 32 + koff];
    pu.hv[1] = *(const v8bf*)&plds[l16 * 32 + koff + 16];
    v16bf pf = pu.v;

    // ctx(16x128) += P(16x32) @ V(32x128): 8 WMMAs over d tiles of 16
#pragma unroll
    for (int g = 0; g < 8; ++g) {
      v16bf bv = load_frag_b(Vh, g * 16, key0, S_LEN);
      acc[g] = wmma_bf16(pf, bv, acc[g]);
    }
    __syncthreads();
  }

  // normalize and store ctx to [s][h*HD + d] bf16
#pragma unroll
  for (int r = 0; r < 8; ++r) {
    int m = r + hgrp * 8;
    float inv = 1.0f / lrow[r];
    bf16_t* orow = Ctx + (size_t)(q0 + m) * (NH_Q * HEAD_DIM)
                       + (size_t)h * HEAD_DIM;
#pragma unroll
    for (int g = 0; g < 8; ++g)
      orow[g * 16 + l16] = (bf16_t)(acc[g][r] * inv);
  }
}

// --------------------------- workspace layout ------------------------------

static constexpr size_t OFF_XB   = 0;                                   // 16 MB
static constexpr size_t OFF_QWB  = OFF_XB   + (size_t)S_LEN * D_MODEL * 2;
static constexpr size_t OFF_KWB  = OFF_QWB  + (size_t)D_MODEL * D_MODEL * 2;
static constexpr size_t OFF_VWB  = OFF_KWB  + (size_t)NH_KV * HEAD_DIM * D_MODEL * 2;
static constexpr size_t OFF_OWB  = OFF_VWB  + (size_t)NH_KV * HEAD_DIM * D_MODEL * 2;
static constexpr size_t OFF_QF   = OFF_OWB  + (size_t)D_MODEL * D_MODEL * 2;
static constexpr size_t OFF_KF   = OFF_QF   + (size_t)S_LEN * D_MODEL * 4;
static constexpr size_t OFF_VF   = OFF_KF   + (size_t)S_LEN * NH_KV * HEAD_DIM * 4;
static constexpr size_t OFF_QB   = OFF_VF   + (size_t)S_LEN * NH_KV * HEAD_DIM * 4;
static constexpr size_t OFF_KB   = OFF_QB   + (size_t)S_LEN * D_MODEL * 2;
static constexpr size_t OFF_VT   = OFF_KB   + (size_t)S_LEN * NH_KV * HEAD_DIM * 2;
static constexpr size_t OFF_CTX  = OFF_VT   + (size_t)S_LEN * NH_KV * HEAD_DIM * 2;

// --------------------------- launcher --------------------------------------

extern "C" void kernel_launch(void* const* d_in, const int* in_sizes, int n_in,
                              void* d_out, int out_size, void* d_ws, size_t ws_size,
                              hipStream_t stream) {
  const float* hs   = (const float*)d_in[0];  // hidden_states [S, D]
  const float* fcos = (const float*)d_in[1];  // [S, HD/2]
  const float* fsin = (const float*)d_in[2];  // [S, HD/2]
  // d_in[3] = attention_mask: unused (causal mask computed analytically)
  const float* qW   = (const float*)d_in[4];  // [H*HD, D]
  const float* kW   = (const float*)d_in[5];  // [KVH*HD, D]
  const float* vW   = (const float*)d_in[6];  // [KVH*HD, D]
  const float* oW   = (const float*)d_in[7];  // [D, H*HD]

  char* w = (char*)d_ws;
  bf16_t* Xb   = (bf16_t*)(w + OFF_XB);
  bf16_t* qWb  = (bf16_t*)(w + OFF_QWB);
  bf16_t* kWb  = (bf16_t*)(w + OFF_KWB);
  bf16_t* vWb  = (bf16_t*)(w + OFF_VWB);
  bf16_t* oWb  = (bf16_t*)(w + OFF_OWB);
  float*  Qf   = (float*) (w + OFF_QF);
  float*  Kf   = (float*) (w + OFF_KF);
  float*  Vf   = (float*) (w + OFF_VF);
  bf16_t* Qb   = (bf16_t*)(w + OFF_QB);
  bf16_t* Kb   = (bf16_t*)(w + OFF_KB);
  bf16_t* Vt   = (bf16_t*)(w + OFF_VT);
  bf16_t* Ctxb = (bf16_t*)(w + OFF_CTX);

  auto cvt = [&](const float* s, bf16_t* d, size_t n) {
    int blocks = (int)((n + 255) / 256);
    f32_to_bf16_kernel<<<blocks, 256, 0, stream>>>(s, d, n);
  };

  // 1. bf16 conversions
  cvt(hs, Xb, (size_t)S_LEN * D_MODEL);
  cvt(qW, qWb, (size_t)D_MODEL * D_MODEL);
  cvt(kW, kWb, (size_t)NH_KV * HEAD_DIM * D_MODEL);
  cvt(vW, vWb, (size_t)NH_KV * HEAD_DIM * D_MODEL);
  cvt(oW, oWb, (size_t)D_MODEL * D_MODEL);

  // 2. QKV projections (bf16 WMMA + async-LDS staging, f32 out)
  gemm_bf16_kernel<<<dim3(D_MODEL / 128, S_LEN / 64), 256, 0, stream>>>(
      Xb, qWb, Qf, S_LEN, D_MODEL, D_MODEL);
  gemm_bf16_kernel<<<dim3((NH_KV * HEAD_DIM) / 128, S_LEN / 64), 256, 0, stream>>>(
      Xb, kWb, Kf, S_LEN, NH_KV * HEAD_DIM, D_MODEL);
  gemm_bf16_kernel<<<dim3((NH_KV * HEAD_DIM) / 128, S_LEN / 64), 256, 0, stream>>>(
      Xb, vWb, Vf, S_LEN, NH_KV * HEAD_DIM, D_MODEL);

  // 3. RoPE + L2-norm -> bf16 head-major
  rope_l2norm_kernel<<<dim3(S_LEN, NH_Q), 64, 0, stream>>>(Qf, fcos, fsin, Qb, NH_Q);
  rope_l2norm_kernel<<<dim3(S_LEN, NH_KV), 64, 0, stream>>>(Kf, fcos, fsin, Kb, NH_KV);

  // 4. V transpose -> [kh][d][s] bf16
  {
    size_t n = (size_t)S_LEN * NH_KV * HEAD_DIM;
    v_transpose_kernel<<<(int)((n + 255) / 256), 256, 0, stream>>>(Vf, Vt);
  }

  // 5. Flash attention
  attn_kernel<<<dim3(S_LEN / 16, NH_Q), 32, 0, stream>>>(Qb, Kb, Vt, Ctxb);

  // 6. Output projection -> f32 d_out
  gemm_bf16_kernel<<<dim3(D_MODEL / 128, S_LEN / 64), 256, 0, stream>>>(
      Ctxb, oWb, (float*)d_out, S_LEN, D_MODEL, D_MODEL);
}
